// SRU_65738769433396
// MI455X (gfx1250) — matrix operand order repaired and checked by reading, not compiled, not run on hardware
//
#include <hip/hip_runtime.h>
#include <hip/hip_bf16.h>
#include <cstdint>
#include <cstddef>

// ---------------- problem constants ----------------
#define LSEQ 2048
#define BB   16
#define DD   1024
#define LB   (LSEQ * BB)     // 32768  (GEMM M)
#define NN   (3 * DD)        // 3072   (GEMM N)

typedef __attribute__((ext_vector_type(16))) __bf16   v16bf;
typedef __attribute__((ext_vector_type(4)))  __bf16   v4bf;
typedef __attribute__((ext_vector_type(8)))  float    v8f;
typedef __attribute__((ext_vector_type(4)))  uint32_t u32x4;

// ---------------- fp32 -> bf16 (RNE) ----------------
__device__ __forceinline__ __bf16 f32_to_bf16(float f) {
    union { float f; uint32_t u; } c; c.f = f;
    uint32_t lsb = (c.u >> 16) & 1u;
    uint32_t r = (c.u + 0x7FFFu + lsb) >> 16;
    union { uint16_t u; __bf16 b; } o; o.u = (uint16_t)r;
    return o.b;
}

__global__ void convert_bf16_kernel(const float* __restrict__ in,
                                    __bf16* __restrict__ out, int n4) {
    int i = blockIdx.x * blockDim.x + threadIdx.x;
    if (i >= n4) return;
    int idx = i * 4;
    float4 v = *(const float4*)(in + idx);
    v4bf o;
    o.x = f32_to_bf16(v.x);
    o.y = f32_to_bf16(v.y);
    o.z = f32_to_bf16(v.z);
    o.w = f32_to_bf16(v.w);
    *(v4bf*)(out + idx) = o;
}

// ---------------- WMMA GEMM + bias + sigmoid epilogue ----------------
// C[m,n] = sum_k A[m,k] * W[n,k]   (A: 32768x1024 bf16, W: 3072x1024 bf16)
// Block: 256 threads = 8 waves. Block tile 128(M) x 128(N).
// K staged in 64-wide chunks, double-buffered LDS filled by
// GLOBAL_LOAD_ASYNC_TO_LDS_B128 (ASYNCcnt path) while WMMAs consume the
// other buffer. Wave (wm 0..3, wn 0..1) owns 32x64 = 2x4 grid of 16x16 tiles.
#define TM 128
#define TN 128
#define TK 64
#define LDK 72            // 64 + 8 pad -> 144-byte rows, 16B aligned, conflict-free

// one 64-byte strip: memory -> LDS, per-lane, async (GVS addressing mode)
__device__ __forceinline__ void async_copy64(uint32_t lds, uint32_t voff,
                                             const void* sbase) {
    asm volatile(
        "global_load_async_to_lds_b128 %0, %1, %2\n\t"
        "global_load_async_to_lds_b128 %0, %1, %2 offset:16\n\t"
        "global_load_async_to_lds_b128 %0, %1, %2 offset:32\n\t"
        "global_load_async_to_lds_b128 %0, %1, %2 offset:48"
        :: "v"(lds), "v"(voff), "s"(sbase) : "memory");
}

__device__ __forceinline__ void wait_async_barrier() {
    asm volatile("s_wait_asynccnt 0x0" ::: "memory");
    __syncthreads();
}

__device__ __forceinline__ void mma_stage(const __bf16* __restrict__ asb,
                                          const __bf16* __restrict__ bsb,
                                          int wm, int wn, int hrow,
                                          int akb, int bkb,
                                          v8f (&acc)[2][4]) {
#pragma unroll
    for (int kk = 0; kk < TK; kk += 32) {
        v16bf afrag[2];
#pragma unroll
        for (int mi = 0; mi < 2; ++mi) {
            union { u32x4 u[2]; v16bf v; } fa;
            const __bf16* ap = asb + (wm * 32 + mi * 16 + hrow) * LDK + kk;
            fa.u[0] = *(const u32x4*)(ap + akb);        // K = kb .. kb+7
            fa.u[1] = *(const u32x4*)(ap + akb + 16);   // K = kb+16 .. kb+23
            afrag[mi] = fa.v;
        }
        v16bf bfrag[4];
#pragma unroll
        for (int ni = 0; ni < 4; ++ni) {
            union { u32x4 u[2]; v16bf v; } fb;
            const __bf16* bp = bsb + (wn * 64 + ni * 16 + hrow) * LDK + kk;
            fb.u[0] = *(const u32x4*)(bp + bkb);        // K = kb .. kb+7
            fb.u[1] = *(const u32x4*)(bp + bkb + 16);   // K = kb+16 .. kb+23
            bfrag[ni] = fb.v;
        }
#pragma unroll
        for (int mi = 0; mi < 2; ++mi)
#pragma unroll
            for (int ni = 0; ni < 4; ++ni)
                acc[mi][ni] = __builtin_amdgcn_wmma_f32_16x16x32_bf16(
                    false, afrag[mi], false, bfrag[ni],
                    (short)0, acc[mi][ni], false, false);
    }
}

__launch_bounds__(256)
__global__ void gemm_gates_kernel(const __bf16* __restrict__ A,
                                  const __bf16* __restrict__ Wb,
                                  const float*  __restrict__ bias,
                                  float* __restrict__ xbuf,
                                  float* __restrict__ zbuf,
                                  float* __restrict__ hbuf) {
    __shared__ __align__(16) __bf16 As[2][TM][LDK];
    __shared__ __align__(16) __bf16 Bs[2][TN][LDK];

    const int tid  = threadIdx.x;
    const int lane = tid & 31;
    const int wave = tid >> 5;     // 0..7
    const int wm   = wave & 3;     // M sub-offset wm*32
    const int wn   = wave >> 2;    // N sub-offset wn*64

    const int m0   = blockIdx.x * TM;
    const int nblk = blockIdx.y;   // 0..23
    const int n0   = nblk * TN;

    v8f acc[2][4];
#pragma unroll
    for (int mi = 0; mi < 2; ++mi)
#pragma unroll
        for (int ni = 0; ni < 4; ++ni)
            acc[mi][ni] = (v8f)(0.0f);

    // cooperative async staging: 2 threads per row, 32 bf16 (64B) each
    const int lrow = tid >> 1;            // 0..127
    const int lks  = (tid & 1) * 32;      // 0 or 32 (elements)

    const uint32_t aoff = (uint32_t)(((m0 + lrow) * DD + lks) * 2);   // bytes
    const uint32_t boff = (uint32_t)(((n0 + lrow) * DD + lks) * 2);   // bytes

    // LDS byte addresses: low 32 bits of flat address == LDS address
    const uint32_t lA0 = (uint32_t)(uintptr_t)&As[0][lrow][lks];
    const uint32_t lA1 = (uint32_t)(uintptr_t)&As[1][lrow][lks];
    const uint32_t lB0 = (uint32_t)(uintptr_t)&Bs[0][lrow][lks];
    const uint32_t lB1 = (uint32_t)(uintptr_t)&Bs[1][lrow][lks];

    const int hrow = lane & 15;               // row/col within 16x16 tile
    const int akb  = (lane < 16) ? 0 : 8;     // A-frag K origin per ISA layout
    const int bkb  = (lane < 16) ? 0 : 8;     // B-frag K origin per ISA layout

    // prologue: fill buffer 0 with k=0 chunk
    async_copy64(lA0, aoff, A);
    async_copy64(lB0, boff, Wb);
    wait_async_barrier();

    for (int k0 = 0; k0 < DD; k0 += 2 * TK) {
        // stage 0: prefetch k0+TK into buf1, compute from buf0 (k0)
        async_copy64(lA1, aoff + (uint32_t)((k0 + TK) * 2), A);
        async_copy64(lB1, boff + (uint32_t)((k0 + TK) * 2), Wb);
        mma_stage(&As[0][0][0], &Bs[0][0][0], wm, wn, hrow, akb, bkb, acc);
        wait_async_barrier();

        // stage 1: prefetch k0+2*TK into buf0, compute from buf1 (k0+TK)
        if (k0 + 2 * TK < DD) {
            async_copy64(lA0, aoff + (uint32_t)((k0 + 2 * TK) * 2), A);
            async_copy64(lB0, boff + (uint32_t)((k0 + 2 * TK) * 2), Wb);
        }
        mma_stage(&As[1][0][0], &Bs[1][0][0], wm, wn, hrow, akb, bkb, acc);
        wait_async_barrier();
    }

    // ---- epilogue: bias, fast sigmoid (v_rcp_f32), store to x/z/h third ----
    float* dst;
    bool gate;
    if (nblk < 8)       { dst = xbuf; gate = false; }
    else if (nblk < 16) { dst = zbuf; gate = true;  }
    else                { dst = hbuf; gate = true;  }
    const int nl0 = n0 & (DD - 1);     // n0 % 1024

#pragma unroll
    for (int ni = 0; ni < 4; ++ni) {
        const int nglob = n0 + wn * 64 + ni * 16 + hrow;
        const float bv = bias[nglob];
        const int col = nl0 + wn * 64 + ni * 16 + hrow;
#pragma unroll
        for (int mi = 0; mi < 2; ++mi) {
            const int rbase = m0 + wm * 32 + mi * 16 + ((lane >> 4) << 3);
#pragma unroll
            for (int r = 0; r < 8; ++r) {
                float v = acc[mi][ni][r] + bv;
                if (gate) {
                    float e = __expf(-v);
                    v = __builtin_amdgcn_rcpf(1.0f + e);   // v_rcp_f32
                }
                dst[(size_t)(rbase + r) * DD + col] = v;
            }
        }
    }
}

// ---------------- sequential scan + highway combine ----------------
// one thread per (b,d): 16384 threads = 512 wave32s, fully coalesced streams
__launch_bounds__(256)
__global__ void sru_scan_kernel(const float* __restrict__ prev,
                                const float* __restrict__ hidden,
                                const float* __restrict__ xbuf,
                                const float* __restrict__ zbuf,
                                const float* __restrict__ hbuf,
                                float* __restrict__ out,
                                float* __restrict__ hfinal) {
    const int idx = blockIdx.x * blockDim.x + threadIdx.x;   // 0..16383
    float s = hidden[idx];
    size_t base = (size_t)idx;
    const size_t stride = (size_t)BB * DD;
#pragma unroll 4
    for (int l = 0; l < LSEQ; ++l, base += stride) {
        const float z = zbuf[base];
        const float x = xbuf[base];
        const float h = hbuf[base];
        const float p = prev[base];
        s = fmaf(z, x - s, s);            // (1-z)*s + z*x
        out[base] = fmaf(h, p - s, s);    // (1-h)*s + h*p
    }
    hfinal[idx] = s;
}

// ---------------- launcher ----------------
extern "C" void kernel_launch(void* const* d_in, const int* in_sizes, int n_in,
                              void* d_out, int out_size, void* d_ws, size_t ws_size,
                              hipStream_t stream) {
    const float* prev   = (const float*)d_in[0];   // (L,B,D)
    const float* hidden = (const float*)d_in[1];   // (B,D)
    const float* W      = (const float*)d_in[2];   // (3D,D)
    const float* bias   = (const float*)d_in[3];   // (3D,)

    float* out    = (float*)d_out;                       // L*B*D
    float* hfinal = out + (size_t)LSEQ * BB * DD;        // + B*D

    // workspace layout
    char* ws = (char*)d_ws;
    __bf16* Ab = (__bf16*)ws;                                    // 64 MB
    __bf16* Wb = (__bf16*)(ws + ((size_t)64 << 20));             //  6 MB
    float* xbuf = (float*)(ws + ((size_t)72 << 20));             // 128 MB
    float* zbuf = xbuf + (size_t)LB * DD;                        // 128 MB
    float* hbuf = zbuf + (size_t)LB * DD;                        // 128 MB

    // 1) fp32 -> bf16 for A and W
    {
        int n4 = (LB * DD) / 4;
        convert_bf16_kernel<<<(n4 + 255) / 256, 256, 0, stream>>>(prev, Ab, n4);
    }
    {
        int n4 = (NN * DD) / 4;
        convert_bf16_kernel<<<(n4 + 255) / 256, 256, 0, stream>>>(W, Wb, n4);
    }

    // 2) WMMA GEMM with fused bias + sigmoid, split into x/z/h buffers
    {
        dim3 grid(LB / TM, NN / TN);   // 256 x 24
        gemm_gates_kernel<<<grid, 256, 0, stream>>>(Ab, Wb, bias, xbuf, zbuf, hbuf);
    }

    // 3) recurrence + highway output
    sru_scan_kernel<<<(BB * DD) / 256, 256, 0, stream>>>(prev, hidden, xbuf, zbuf,
                                                         hbuf, out, hfinal);
}
